// Net_20057497272472
// MI455X (gfx1250) — matrix operand order repaired and checked by reading, not compiled, not run on hardware
//
#include <hip/hip_runtime.h>

typedef __attribute__((ext_vector_type(16))) __bf16 v16bf;
typedef __attribute__((ext_vector_type(8)))  __bf16 v8bf;
typedef __attribute__((ext_vector_type(8)))  float  v8f;

#define B_ 256
#define T_ 2048
#define D_ 32
#define H_ 128
#define HS 136   /* padded lds_h row stride (bf16 elems): 272B -> conflict-free A-pack */
#define XS 40    /* padded lds_x row stride */

__device__ __forceinline__ float sigmoid_f(float x) {
  return __builtin_amdgcn_rcpf(1.0f + __expf(-x));
}
__device__ __forceinline__ float tanh_f(float x) {
#if __has_builtin(__builtin_amdgcn_tanhf)
  return __builtin_amdgcn_tanhf(x);
#else
  return 1.0f - 2.0f * __builtin_amdgcn_rcpf(1.0f + __expf(2.0f * x));
#endif
}

// A-matrix pack (16-bit, 16x32): low lanes hold K=kb..kb+7 and kb+16..kb+23
__device__ __forceinline__ v16bf pack_a(const __bf16* p) {
  const v8bf lo = *(const v8bf*)p;
  const v8bf hi = *(const v8bf*)(p + 16);
  v16bf r;
#pragma unroll
  for (int e = 0; e < 8; ++e) { r[e] = lo[e]; r[e + 8] = hi[e]; }
  return r;
}

// B-matrix pack (16-bit, 32x16): lane L holds n=L&15, K=kb..kb+7 and kb+16..kb+23 (kb=8*(L>=16))
__device__ __forceinline__ v16bf pack_b_f32(const float* p) {
  v16bf r;
#pragma unroll
  for (int e = 0; e < 8; ++e) { r[e] = (__bf16)p[e]; r[e + 8] = (__bf16)p[e + 16]; }
  return r;
}

__global__ __launch_bounds__(256, 1)
void lstm_wmma_kernel(const float* __restrict__ points, const float* __restrict__ times,
                      const float* __restrict__ W_ih, const float* __restrict__ W_hh,
                      const float* __restrict__ b_ih, const float* __restrict__ b_hh,
                      const float* __restrict__ W_lin, const float* __restrict__ b_lin,
                      float* __restrict__ out)
{
  __shared__ __align__(16) __bf16 lds_h[16 * HS];
  __shared__ __align__(16) __bf16 lds_x[16 * XS];
  __shared__ float lds_t[16];

  const int tid  = threadIdx.x;
  const int lane = tid & 31;
  const int wid  = tid >> 5;       // 8 waves
  const int half = lane >> 4;
  const int l15  = lane & 15;
  const int j    = wid * 16;       // this wave's h-column tile
  const int b0   = blockIdx.x * 16;

  // ---- preload weights into VGPRs in WMMA B-layout (bf16) ----
  v16bf Bhh[4][4];   // [gate][k-chunk]  : 128 VGPRs
  v16bf Bih[4];      // [gate], K=0..31  :  32 VGPRs
  float wt[4], bv[4];
#pragma unroll
  for (int g = 0; g < 4; ++g) {
    const int n = g * H_ + j + l15;                 // gate output column
    bv[g] = b_ih[n] + b_hh[n];
    wt[g] = W_ih[n * (D_ + 1) + D_];                // time-column weight (rank-1 term)
    Bih[g] = pack_b_f32(W_ih + n * (D_ + 1) + half * 8);
#pragma unroll
    for (int kc = 0; kc < 4; ++kc)
      Bhh[g][kc] = pack_b_f32(W_hh + n * H_ + kc * 32 + half * 8);
  }

  // zero initial hidden state in LDS
  for (int i = tid; i < 16 * HS; i += 256) lds_h[i] = (__bf16)0.0f;

  // global-x prefetch registers for t=0 (2 floats / thread)
  const int xrow = tid >> 4;
  const int xcol = (tid & 15) * 2;
  const float* xg = points + (size_t)(b0 + xrow) * T_ * D_ + xcol;
  float2 xr = *(const float2*)xg;
  float  tr = times[(size_t)(b0 + (tid & 15)) * T_];

  v8f cst;   // cell state tile, fp32, wave-local
#pragma unroll
  for (int r = 0; r < 8; ++r) cst[r] = 0.0f;

  __syncthreads();

#pragma unroll 1
  for (int t = 0; t < T_; ++t) {
    // stage x_t (bf16) + times into LDS, then prefetch x_{t+1}
    lds_x[xrow * XS + xcol]     = (__bf16)xr.x;
    lds_x[xrow * XS + xcol + 1] = (__bf16)xr.y;
    if (tid < 16) lds_t[tid] = tr;
    if (t + 1 < T_) {
      xr = *(const float2*)(xg + (size_t)(t + 1) * D_);
      tr = times[(size_t)(b0 + (tid & 15)) * T_ + t + 1];
    }
    __syncthreads();

    // pack A operands from LDS (ds_load_b128 pairs, conflict-free via padding)
    const v16bf Ax = pack_a(&lds_x[l15 * XS + half * 8]);
    v16bf Ah[4];
#pragma unroll
    for (int kc = 0; kc < 4; ++kc)
      Ah[kc] = pack_a(&lds_h[l15 * HS + kc * 32 + half * 8]);
    float tv[8];
#pragma unroll
    for (int r = 0; r < 8; ++r) tv[r] = lds_t[half * 8 + r];

    // gates = bias + t*w_t + x@Wih^T + h@Whh^T   (20 WMMAs / wave / step)
    v8f acc[4];
#pragma unroll
    for (int g = 0; g < 4; ++g) {
#pragma unroll
      for (int r = 0; r < 8; ++r) acc[g][r] = bv[g] + tv[r] * wt[g];
      acc[g] = __builtin_amdgcn_wmma_f32_16x16x32_bf16(
          false, Ax, false, Bih[g], (short)0, acc[g], false, false);
#pragma unroll
      for (int kc = 0; kc < 4; ++kc)
        acc[g] = __builtin_amdgcn_wmma_f32_16x16x32_bf16(
            false, Ah[kc], false, Bhh[g][kc], (short)0, acc[g], false, false);
    }

    // LSTM cell (gate order i,f,g,o)
    __bf16 hv[8];
#pragma unroll
    for (int r = 0; r < 8; ++r) {
      const float iv = sigmoid_f(acc[0][r]);
      const float fv = sigmoid_f(acc[1][r]);
      const float gv = tanh_f(acc[2][r]);
      const float ov = sigmoid_f(acc[3][r]);
      const float cc = fv * cst[r] + iv * gv;
      cst[r] = cc;
      hv[r] = (__bf16)(ov * tanh_f(cc));
    }

    __syncthreads();   // all waves done reading old h before overwrite
#pragma unroll
    for (int r = 0; r < 8; ++r)
      lds_h[(half * 8 + r) * HS + j + l15] = hv[r];   // C-layout: n=l15(+j), m=half*8+r
  }

  __syncthreads();

  // final linear (H->2) + log_softmax on last hidden state
  if (tid < 16) {
    const int m = tid;
    float l0 = b_lin[0], l1 = b_lin[1];
    for (int n = 0; n < H_; ++n) {
      const float hvf = (float)lds_h[m * HS + n];
      l0 += hvf * W_lin[n];
      l1 += hvf * W_lin[H_ + n];
    }
    const float mx  = fmaxf(l0, l1);
    const float lse = mx + __logf(__expf(l0 - mx) + __expf(l1 - mx));
    out[(b0 + m) * 2 + 0] = l0 - lse;
    out[(b0 + m) * 2 + 1] = l1 - lse;
  }
}

extern "C" void kernel_launch(void* const* d_in, const int* in_sizes, int n_in,
                              void* d_out, int out_size, void* d_ws, size_t ws_size,
                              hipStream_t stream) {
  (void)in_sizes; (void)n_in; (void)d_ws; (void)ws_size; (void)out_size;
  const float* points = (const float*)d_in[0];
  const float* times  = (const float*)d_in[1];
  const float* W_ih   = (const float*)d_in[2];
  const float* W_hh   = (const float*)d_in[3];
  const float* b_ih   = (const float*)d_in[4];
  const float* b_hh   = (const float*)d_in[5];
  const float* W_lin  = (const float*)d_in[6];
  const float* b_lin  = (const float*)d_in[7];
  lstm_wmma_kernel<<<dim3(B_ / 16), dim3(256), 0, stream>>>(
      points, times, W_ih, W_hh, b_ih, b_hh, W_lin, b_lin, (float*)d_out);
}
